// LGA3D2_15796889715116
// MI455X (gfx1250) — compile-verified, matrix operand order checked by reading, not run on hardware
//
#include <hip/hip_runtime.h>

// LGA3D2 on gfx1250 via v_wmma_f32_16x16x32_f16.
//
// out[c,n] (16 channels x 16 w-columns, fixed b,h,d) =
//   sum_{ik in 5x3} A_ik (16x32 f16) x B_ik (32x16 f16 banded guidance)
// A_ik[c,u] = x[b, c, d+k-1, h+i-2, w0-2+u]  (u=0..19, zero-padded OOB, K pad->32)
// B_ik[u,n] = g[b, (i*5+(u-n))*3+k, h, w0+n] for 0<=u-n<=4, else 0 (d-invariant,
//             built once per block, register-resident per lane).
//
// v2: f32->f16 conversion hoisted out of the hot loop. A prepass converts the
// cost volume to f16 (fits L2); pass1 runs f16->f16, pass2 f16->f32. f16 A
// fragments load as raw dwords straight into the WMMA register layout
// (VGPR v <-> K pair 2v,2v+1), eliminating all per-tap cvt/pack VALU.

typedef __attribute__((ext_vector_type(16))) _Float16     v16h;
typedef __attribute__((ext_vector_type(4)))  _Float16     v4h;
typedef __attribute__((ext_vector_type(8)))  float        v8f;
typedef __attribute__((ext_vector_type(8)))  unsigned int v8u;

#define Bn   2
#define Cc   16
#define Dd   48
#define Hh   128
#define Ww   256
#define Gg   75
#define WT   16    // output w-tile width (WMMA N)
#define NIK  15    // (i,k) tap pairs: 5 spatial rows x 3 depth offsets

// ---- prepass: f32 cost volume -> f16 (vectorized 4-wide) ----
__global__ __launch_bounds__(256)
void cvt_f32_f16(const float4* __restrict__ in, v4h* __restrict__ out)
{
  const size_t i = (size_t)blockIdx.x * 256 + threadIdx.x;
  const float4 v = in[i];
  v4h h; h[0] = (_Float16)v.x; h[1] = (_Float16)v.y;
         h[2] = (_Float16)v.z; h[3] = (_Float16)v.w;
  out[i] = h;
}

template <typename InT, typename OutT>
__global__ __launch_bounds__(256)
void lga3d_pass(const InT* __restrict__ x, const float* __restrict__ g,
                OutT* __restrict__ out)
{
  const int wt   = blockIdx.x;           // w tile index
  const int h    = blockIdx.y;
  const int b    = blockIdx.z;
  const int w0   = wt * WT;
  const int tid  = threadIdx.x;
  const int lane = tid & 31;
  const int wave = tid >> 5;

  __shared__ __align__(16) float     gs[Gg][WT];          // staged guidance
  __shared__ __align__(16) _Float16  gB[NIK][WT][32];     // banded B: [ik][n][u]

  // ---- stage guidance for this (b,h,w-tile) ----
  for (int idx = tid; idx < Gg * WT; idx += 256) {
    const int ch = idx >> 4, n = idx & 15;
    gs[ch][n] = g[(((size_t)b * Gg + ch) * Hh + h) * Ww + w0 + n];
  }
  __syncthreads();

  // ---- build 15 banded 32x16 B matrices (f16), zero outside the band ----
  for (int idx = tid; idx < NIK * WT * 32; idx += 256) {
    const int u  = idx & 31;
    const int n  = (idx >> 5) & 15;
    const int ik = idx >> 9;
    const int i  = ik / 3, k = ik - 3 * i;
    const int j  = u - n;                       // n<=15, j<=4 => u<=19 < 20 always
    _Float16 v = (_Float16)0.f;
    if (j >= 0 && j < 5) v = (_Float16)gs[(i * 5 + j) * 3 + k][n];
    gB[ik][n][u] = v;
  }
  __syncthreads();

  // ---- per-lane fragment geometry ----
  // 16-bit 16x32 A layout (ISA 7.12.2): lanes 0-15 hold K {0..7, 16..23},
  // lanes 16-31 hold K {8..15, 24..31}; element e -> K = (e<8 ? e : 8+e) + hiK.
  // Dense 16-bit 32x16 B assumed to mirror this with column N = lane%16.
  const int nl  = lane & 15;                 // B column / output column n; also A row c
  const int hiK = (lane < 16) ? 0 : 8;       // K-chunk select for this lane half

  v16h Bf[NIK];
  #pragma unroll
  for (int ik = 0; ik < NIK; ++ik) {
    v16h f;
    #pragma unroll
    for (int e = 0; e < 8; ++e) {
      f[e]     = gB[ik][nl][hiK + e];          // K = hiK .. hiK+7
      f[8 + e] = gB[ik][nl][16 + hiK + e];     // K = 16+hiK .. 23+hiK
    }
    Bf[ik] = f;
  }

  const int  c        = nl;                  // A row = channel
  const bool interior = (w0 >= 2) && (w0 + 17 < Ww);
  const int  d0       = wave * (Dd / 8);     // 6 depths per wave

  for (int d = d0; d < d0 + Dd / 8; ++d) {
    v8f acc = {};
    #pragma unroll
    for (int ik = 0; ik < NIK; ++ik) {
      const int i  = ik / 3, k = ik - 3 * i;
      const int dd = d + k - 1;
      const int hh = h + i - 2;
      v16h Af = {};                          // zero => zero-padding contribution
      if (dd >= 0 && dd < Dd && hh >= 0 && hh < Hh) {
        const InT* row   = x + ((((size_t)b * Cc + c) * Dd + dd) * Hh + hh) * Ww;
        const int  wbase = w0 - 2 + hiK;     // u = hiK + e, e = 0..7 (even index)
        if constexpr (sizeof(InT) == 2) {
          // f16 source: dwords map 1:1 onto the fragment (VGPR v = halfs 2v,2v+1)
          if (interior) {
            const unsigned int* rp = (const unsigned int*)(row + wbase);  // 4B aligned
            v8u au = {};
            au[0] = rp[0]; au[1] = rp[1]; au[2] = rp[2]; au[3] = rp[3];
            if (lane < 16) {                 // K = 16..19 <=> u = 16..19
              const unsigned int* rp2 = (const unsigned int*)(row + w0 + 14);
              au[4] = rp2[0]; au[5] = rp2[1];
            }
            Af = __builtin_bit_cast(v16h, au);
          } else {
            #pragma unroll
            for (int e = 0; e < 8; ++e) {
              const int w = wbase + e;
              if (w >= 0 && w < Ww) Af[e] = (_Float16)row[w];
            }
            if (lane < 16) {
              #pragma unroll
              for (int e = 0; e < 4; ++e) {
                const int w = w0 + 14 + e;
                if (w < Ww) Af[8 + e] = (_Float16)row[w];
              }
            }
          }
        } else {
          // f32 source fallback path (converts in-loop)
          if (interior) {
            const float2* p = (const float2*)(row + wbase);                // 8B aligned
            const float2 q0 = p[0], q1 = p[1], q2 = p[2], q3 = p[3];
            Af[0] = (_Float16)q0.x; Af[1] = (_Float16)q0.y;
            Af[2] = (_Float16)q1.x; Af[3] = (_Float16)q1.y;
            Af[4] = (_Float16)q2.x; Af[5] = (_Float16)q2.y;
            Af[6] = (_Float16)q3.x; Af[7] = (_Float16)q3.y;
            if (lane < 16) {
              const float2* p2 = (const float2*)(row + w0 + 14);
              const float2 r0 = p2[0], r1 = p2[1];
              Af[8]  = (_Float16)r0.x; Af[9]  = (_Float16)r0.y;
              Af[10] = (_Float16)r1.x; Af[11] = (_Float16)r1.y;
            }
          } else {
            #pragma unroll
            for (int e = 0; e < 8; ++e) {
              const int w = wbase + e;
              if (w >= 0 && w < Ww) Af[e] = (_Float16)row[w];
            }
            if (lane < 16) {
              #pragma unroll
              for (int e = 0; e < 4; ++e) {
                const int w = w0 + 14 + e;
                if (w < Ww) Af[8 + e] = (_Float16)row[w];
              }
            }
          }
        }
      }
      acc = __builtin_amdgcn_wmma_f32_16x16x32_f16(
                false, Af, false, Bf[ik], (short)0, acc, false, false);
    }
    // ---- store 16x16 tile per C/D layout: VGPR r -> M = mbase + r, N = nl ----
    const int    mbase   = (lane < 16) ? 0 : 8;
    const size_t cstride = (size_t)Dd * Hh * Ww;
    OutT* o = out + ((((size_t)b * Cc + mbase) * Dd + d) * Hh + h) * Ww + w0 + nl;
    #pragma unroll
    for (int r = 0; r < 8; ++r) o[r * cstride] = (OutT)acc[r];
  }
}

extern "C" void kernel_launch(void* const* d_in, const int* in_sizes, int n_in,
                              void* d_out, int out_size, void* d_ws, size_t ws_size,
                              hipStream_t stream)
{
  (void)in_sizes; (void)n_in;
  const float* x   = (const float*)d_in[0];
  const float* g   = (const float*)d_in[1];
  float*       out = (float*)d_out;

  const size_t n = (size_t)out_size;               // B*C*D*H*W elements
  const dim3 grid(Ww / WT, Hh, Bn);
  const dim3 block(256);

  if (ws_size >= n * 4) {
    // f16 fast path: ws = [ xh : n f16 | midh : n f16 ]
    _Float16* xh   = (_Float16*)d_ws;
    _Float16* midh = xh + n;
    cvt_f32_f16<<<dim3((unsigned)(n / 4 / 256)), block, 0, stream>>>(
        (const float4*)x, (v4h*)xh);
    lga3d_pass<_Float16, _Float16><<<grid, block, 0, stream>>>(xh,   g, midh);
    lga3d_pass<_Float16, float   ><<<grid, block, 0, stream>>>(midh, g, out);
  } else {
    // fallback: f32 path, d_out doubles as (imperfect) intermediate
    lga3d_pass<float, float><<<grid, block, 0, stream>>>(x,   g, out);
    lga3d_pass<float, float><<<grid, block, 0, stream>>>(out, g, out);
  }
}
